// LogicalReasoner_37864431681732
// MI455X (gfx1250) — matrix operand order repaired and checked by reading, not compiled, not run on hardware
//
#include <hip/hip_runtime.h>
#include <hip/hip_bf16.h>
#include <math.h>

typedef __bf16 bf16_t;
typedef __attribute__((ext_vector_type(16))) __bf16 v16bf;
typedef __attribute__((ext_vector_type(8)))  __bf16 v8bf;
typedef __attribute__((ext_vector_type(8)))  float  v8f;

#define S_LEN 256
#define H_DIM 384
#define I_DIM 1536
#define NHEAD 8
#define DHEAD 48
#define NLAYER 2
#define NSTEPS 2

__device__ __forceinline__ float gelu_exact(float x) {
  return 0.5f * x * (1.0f + erff(x * 0.7071067811865475f));
}
__device__ __forceinline__ float sigmoid_f(float x) {
  return 1.0f / (1.0f + __expf(-x));
}

// ---------------- WMMA fragment loaders (wave32, 16x16x32 bf16) -------------
// NOTE: no bounds guards -- the launcher only ever uses shapes that are exact
// multiples of the tile (M,N % 64 == 0, K % 32 == 0), so every access is
// in-range and loads vectorize to global_load_b128.

// A fragment (16x32 tile of row-major [rows,K]): lane holds row m0+lane%16;
// element j in 0..15 -> K = k0 + 16*(lane/16) + j
__device__ __forceinline__ v16bf fragA_f32(const float* base, long long ld,
                                           int row0, int k0, int lr, int hi) {
  const float* p = base + (long long)(row0 + lr) * ld + k0 + 16 * hi;
  float4 a0 = *(const float4*)(p);
  float4 a1 = *(const float4*)(p + 4);
  float4 b0 = *(const float4*)(p + 8);
  float4 b1 = *(const float4*)(p + 12);
  v16bf f;
  f[0] = (bf16_t)a0.x;  f[1] = (bf16_t)a0.y;  f[2] = (bf16_t)a0.z;  f[3] = (bf16_t)a0.w;
  f[4] = (bf16_t)a1.x;  f[5] = (bf16_t)a1.y;  f[6] = (bf16_t)a1.z;  f[7] = (bf16_t)a1.w;
  f[8] = (bf16_t)b0.x;  f[9] = (bf16_t)b0.y;  f[10] = (bf16_t)b0.z; f[11] = (bf16_t)b0.w;
  f[12] = (bf16_t)b1.x; f[13] = (bf16_t)b1.y; f[14] = (bf16_t)b1.z; f[15] = (bf16_t)b1.w;
  return f;
}
__device__ __forceinline__ v16bf fragA_bf16(const bf16_t* base, long long ld,
                                            int row0, int k0, int lr, int hi) {
  const bf16_t* p = base + (long long)(row0 + lr) * ld + k0 + 16 * hi;
  v8bf lo = *(const v8bf*)(p);       // 16B
  v8bf hf = *(const v8bf*)(p + 8);   // 16B
  v16bf f;
#pragma unroll
  for (int j = 0; j < 8; ++j) { f[j] = lo[j]; f[j + 8] = hf[j]; }
  return f;
}
// B fragment for C = A @ W^T with W row-major [N,K]: lane holds col n0+lane%16,
// element e -> K = k0 + 16*(lane/16) + e  (B rows striped across lanes)
__device__ __forceinline__ v16bf fragB_w_f32(const float* base, long long ld,
                                             int n0, int k0, int lr, int hi) {
  const float* p = base + (long long)(n0 + lr) * ld + k0 + 16 * hi;
  float4 a0 = *(const float4*)(p);
  float4 a1 = *(const float4*)(p + 4);
  float4 a2 = *(const float4*)(p + 8);
  float4 a3 = *(const float4*)(p + 12);
  v16bf f;
  f[0] = (bf16_t)a0.x;  f[1] = (bf16_t)a0.y;  f[2] = (bf16_t)a0.z;  f[3] = (bf16_t)a0.w;
  f[4] = (bf16_t)a1.x;  f[5] = (bf16_t)a1.y;  f[6] = (bf16_t)a1.z;  f[7] = (bf16_t)a1.w;
  f[8] = (bf16_t)a2.x;  f[9] = (bf16_t)a2.y;  f[10] = (bf16_t)a2.z; f[11] = (bf16_t)a2.w;
  f[12] = (bf16_t)a3.x; f[13] = (bf16_t)a3.y; f[14] = (bf16_t)a3.z; f[15] = (bf16_t)a3.w;
  return f;
}
// B fragment for C = A @ B with B row-major [K,N] (K-major transpose gather)
__device__ __forceinline__ v16bf fragB_km_f32(const float* base, long long ld,
                                              int n0, int k0, int lr, int hi) {
  const float* p = base + (long long)(k0 + 16 * hi) * ld + (n0 + lr);
  v16bf f;
#pragma unroll
  for (int e = 0; e < 16; ++e) f[e] = (bf16_t)p[(long long)e * ld];
  return f;
}

__device__ __forceinline__ v8f wmma_bf16(v16bf a, v16bf b, v8f c) {
  return __builtin_amdgcn_wmma_f32_16x16x32_bf16(false, a, false, b,
                                                 (short)0, c, false, false);
}

// ---------------- generic WMMA GEMM ----------------------------------------
// C[m,n] (+= bias)(-> gelu) = sum_k A[m,k] * B'[k,n]
// BKMAJ=0: B is W-style row-major [N,K];  BKMAJ=1: B is row-major [K,N]
// block = 256 threads = 8 waves; block tile 64(M) x 64(N); wave tile 16x32
template <int BKMAJ, int ABF16, int GELUE, int OUTBF16, int BIASM, int SCAT>
__global__ __launch_bounds__(256) void gemm_wmma_k(
    const void* __restrict__ Ap, long long lda,
    const void* __restrict__ Bp, long long ldb, long long zStrideB,
    const float* __restrict__ bias,
    void* __restrict__ Cp, long long ldc, long long zStrideC,
    int K) {
  int w = threadIdx.x >> 5;
  int lane = threadIdx.x & 31;
  int lr = lane & 15, hi = (lane >> 4) & 1;
  int m0 = blockIdx.x * 64 + (w & 3) * 16;
  int n0 = blockIdx.y * 64 + (w >> 2) * 32;
  const float*  Af = (const float*)Ap;
  const bf16_t* Ab = (const bf16_t*)Ap;
  const float*  Bf = (const float*)Bp + (long long)blockIdx.z * zStrideB;

  v8f acc0 = {};
  v8f acc1 = {};
  for (int k0 = 0; k0 < K; k0 += 32) {
    v16bf a;
    if (ABF16) a = fragA_bf16(Ab, lda, m0, k0, lr, hi);
    else       a = fragA_f32(Af, lda, m0, k0, lr, hi);
    v16bf b0, b1;
    if (BKMAJ) {
      b0 = fragB_km_f32(Bf, ldb, n0,      k0, lr, hi);
      b1 = fragB_km_f32(Bf, ldb, n0 + 16, k0, lr, hi);
    } else {
      b0 = fragB_w_f32(Bf, ldb, n0,      k0, lr, hi);
      b1 = fragB_w_f32(Bf, ldb, n0 + 16, k0, lr, hi);
    }
    acc0 = wmma_bf16(a, b0, acc0);
    acc1 = wmma_bf16(a, b1, acc1);
  }
#pragma unroll
  for (int half = 0; half < 2; ++half) {
    v8f acc = half ? acc1 : acc0;
    int nn0 = n0 + 16 * half;
#pragma unroll
    for (int r = 0; r < 8; ++r) {
      int m = m0 + r + 8 * hi;
      int n = nn0 + lr;
      float v = acc[r];
      if (bias) v += BIASM ? bias[m % H_DIM] : bias[n];
      if (GELUE) v = gelu_exact(v);
      long long addr;
      if (SCAT) {  // write inter as [i, j, o]: m = i*H+o, n = j
        long long i = m / H_DIM, o = m % H_DIM;
        addr = (i * S_LEN + n) * H_DIM + o;
      } else {
        addr = (long long)m * ldc + n + (long long)blockIdx.z * zStrideC;
      }
      if (OUTBF16) ((bf16_t*)Cp)[addr] = (bf16_t)v;
      else         ((float*)Cp)[addr]  = v;
    }
  }
}

// ---------- fused contradiction head: sigmoid(gelu(inter@Wc1^T+bc1)@Wc2+bc2)
// inter: bf16 [S*S, H] rows (i,j), K = o. Each wave: 16 rows, sweep all 24
// n-tiles, reduce gelu(tile)*Wc2[n] across lanes via LDS ds_add_f32.
__global__ __launch_bounds__(256) void contra_head_k(
    const bf16_t* __restrict__ interB,
    const float* __restrict__ Wc1, const float* __restrict__ bc1,
    const float* __restrict__ Wc2, const float* __restrict__ bc2,
    float* __restrict__ probs) {
  __shared__ float red[8][16];
  int w = threadIdx.x >> 5;
  int lane = threadIdx.x & 31;
  int lr = lane & 15, hi = (lane >> 4) & 1;
  int m0 = (blockIdx.x * 8 + w) * 16;
  if (lane < 16) red[w][lane] = 0.0f;
  __syncthreads();
  float pm[8];
#pragma unroll
  for (int r = 0; r < 8; ++r) pm[r] = 0.0f;
  for (int nt = 0; nt < H_DIM / 16; ++nt) {
    int n0 = nt * 16;
    v8f acc = {};
    for (int k0 = 0; k0 < H_DIM; k0 += 32) {
      v16bf a = fragA_bf16(interB, H_DIM, m0, k0, lr, hi);
      v16bf b = fragB_w_f32(Wc1, H_DIM, n0, k0, lr, hi);
      acc = wmma_bf16(a, b, acc);
    }
    int n = n0 + lr;
    float bb = bc1[n], w2 = Wc2[n];
#pragma unroll
    for (int r = 0; r < 8; ++r) pm[r] += gelu_exact(acc[r] + bb) * w2;
  }
#pragma unroll
  for (int r = 0; r < 8; ++r) atomicAdd(&red[w][r + 8 * hi], pm[r]);
  __syncthreads();
  if (lane < 16) probs[m0 + lane] = sigmoid_f(red[w][lane] + bc2[0]);
}

// ---------------- attention (scalar; <1% of FLOPs) --------------------------
__global__ __launch_bounds__(256) void attention_k(const float* __restrict__ qkv,
                                                   float* __restrict__ out) {
  __shared__ float qs[DHEAD];
  __shared__ float prob[S_LEN];
  __shared__ float red[256];
  int i = blockIdx.x, h = blockIdx.y, tid = threadIdx.x;
  const float scale = 0.14433756729740645f;  // 1/sqrt(48)
  if (tid < DHEAD) qs[tid] = qkv[(long long)i * (3 * H_DIM) + h * DHEAD + tid] * scale;
  __syncthreads();
  const float* krow = qkv + (long long)tid * (3 * H_DIM) + H_DIM + h * DHEAD;
  float s = 0.0f;
#pragma unroll
  for (int d = 0; d < DHEAD; ++d) s += qs[d] * krow[d];
  red[tid] = s;
  __syncthreads();
  for (int off = 128; off > 0; off >>= 1) {
    if (tid < off) red[tid] = fmaxf(red[tid], red[tid + off]);
    __syncthreads();
  }
  float mx = red[0];
  __syncthreads();
  float e = __expf(s - mx);
  prob[tid] = e;
  red[tid] = e;
  __syncthreads();
  for (int off = 128; off > 0; off >>= 1) {
    if (tid < off) red[tid] += red[tid + off];
    __syncthreads();
  }
  float inv = 1.0f / red[0];
  __syncthreads();
  if (tid < DHEAD) {
    float acc = 0.0f;
    for (int j = 0; j < S_LEN; ++j)
      acc += prob[j] * qkv[(long long)j * (3 * H_DIM) + 2 * H_DIM + h * DHEAD + tid];
    out[(long long)i * H_DIM + h * DHEAD + tid] = acc * inv;
  }
}

// ---------------- residual add + LayerNorm (in place on x) ------------------
__global__ __launch_bounds__(256) void add_ln_k(float* __restrict__ x,
                                                const float* __restrict__ res,
                                                const float* __restrict__ g,
                                                const float* __restrict__ b) {
  __shared__ float s1[256], s2[256];
  int row = blockIdx.x, tid = threadIdx.x;
  float* xr = x + (long long)row * H_DIM;
  const float* rr = res + (long long)row * H_DIM;
  float a0 = xr[tid] + rr[tid];
  int h2 = tid + 256;
  bool has2 = h2 < H_DIM;
  float a1 = has2 ? (xr[h2] + rr[h2]) : 0.0f;
  s1[tid] = a0 + a1;
  s2[tid] = a0 * a0 + a1 * a1;
  __syncthreads();
  for (int off = 128; off > 0; off >>= 1) {
    if (tid < off) { s1[tid] += s1[tid + off]; s2[tid] += s2[tid + off]; }
    __syncthreads();
  }
  float mean = s1[0] * (1.0f / H_DIM);
  float var = s2[0] * (1.0f / H_DIM) - mean * mean;
  float rstd = rsqrtf(var + 1e-5f);
  __syncthreads();
  xr[tid] = (a0 - mean) * rstd * g[tid] + b[tid];
  if (has2) xr[h2] = (a1 - mean) * rstd * g[h2] + b[h2];
}

// ---------------- small utility kernels -------------------------------------
__global__ void copy_k(float* dst, const float* src, int n) {
  int i = blockIdx.x * 256 + threadIdx.x;
  if (i < n) dst[i] = src[i];
}
__global__ void zero_k(float* dst, int n) {
  int i = blockIdx.x * 256 + threadIdx.x;
  if (i < n) dst[i] = 0.0f;
}
__global__ void meanpool_k(const float* __restrict__ x, float* __restrict__ pooled) {
  int h = blockIdx.x * 256 + threadIdx.x;
  if (h < H_DIM) {
    float s = 0.0f;
    for (int i = 0; i < S_LEN; ++i) s += x[(long long)i * H_DIM + h];
    pooled[h] = s * (1.0f / S_LEN);
  }
}
__global__ void gemv_k(const float* __restrict__ v, const float* __restrict__ W,
                       const float* __restrict__ b, float* __restrict__ y,
                       int N, int K, int dogelu) {
  int n = blockIdx.x * 256 + threadIdx.x;
  if (n < N) {
    float s = b ? b[n] : 0.0f;
    const float* wr = W + (long long)n * K;
    for (int k = 0; k < K; ++k) s += v[k] * wr[k];
    y[n] = dogelu ? gelu_exact(s) : s;
  }
}
__global__ void gru_k(float* __restrict__ rstate, const float* __restrict__ gi,
                      const float* __restrict__ gh) {
  int h = blockIdx.x * 256 + threadIdx.x;
  if (h < H_DIM) {
    float r = sigmoid_f(gi[h] + gh[h]);
    float z = sigmoid_f(gi[H_DIM + h] + gh[H_DIM + h]);
    float n = tanhf(gi[2 * H_DIM + h] + r * gh[2 * H_DIM + h]);
    rstate[h] = (1.0f - z) * n + z * rstate[h];
  }
}
__global__ __launch_bounds__(256) void dotsig_k(const float* __restrict__ p1,
                                                const float* __restrict__ Wp2,
                                                const float* __restrict__ bp2,
                                                float* __restrict__ out) {
  __shared__ float red[256];
  int tid = threadIdx.x;
  float s = 0.0f;
  for (int h = tid; h < H_DIM; h += 256) s += p1[h] * Wp2[h];
  red[tid] = s;
  __syncthreads();
  for (int off = 128; off > 0; off >>= 1) {
    if (tid < off) red[tid] += red[tid + off];
    __syncthreads();
  }
  if (tid == 0) out[0] = sigmoid_f(red[0] + bp2[0]);
}

// ---------------- host launcher ---------------------------------------------
extern "C" void kernel_launch(void* const* d_in, const int* in_sizes, int n_in,
                              void* d_out, int out_size, void* d_ws, size_t ws_size,
                              hipStream_t stream) {
  (void)in_sizes; (void)n_in; (void)out_size; (void)ws_size;
  const float* hidden = (const float*)d_in[0];
  const float* Wqkv = (const float*)d_in[1];
  const float* bqkv = (const float*)d_in[2];
  const float* Wo   = (const float*)d_in[3];
  const float* bo   = (const float*)d_in[4];
  const float* W1   = (const float*)d_in[5];
  const float* b1   = (const float*)d_in[6];
  const float* W2   = (const float*)d_in[7];
  const float* b2   = (const float*)d_in[8];
  const float* g1   = (const float*)d_in[9];
  const float* be1  = (const float*)d_in[10];
  const float* g2   = (const float*)d_in[11];
  const float* be2  = (const float*)d_in[12];
  const float* Wbil = (const float*)d_in[13];
  const float* bbil = (const float*)d_in[14];
  const float* Wc1  = (const float*)d_in[15];
  const float* bc1  = (const float*)d_in[16];
  const float* Wc2  = (const float*)d_in[17];
  const float* bc2  = (const float*)d_in[18];
  const float* Wih  = (const float*)d_in[19];
  const float* bih  = (const float*)d_in[20];
  const float* Whh  = (const float*)d_in[21];
  const float* bhh  = (const float*)d_in[22];
  const float* Wp1  = (const float*)d_in[23];
  const float* bp1  = (const float*)d_in[24];
  const float* Wp2  = (const float*)d_in[25];
  const float* bp2  = (const float*)d_in[26];
  const float* Wout = (const float*)d_in[27];
  const float* bout = (const float*)d_in[28];
  // d_in[29] = reasoning_steps (== 2 in setup; cannot read a device scalar
  // during graph capture, so fixed at NSTEPS)

  float* out = (float*)d_out;
  float* outFinal = out;                                // [256*384]
  float* outCprob = out + S_LEN * H_DIM;                // [2][256*256]
  float* outConts = outCprob + NSTEPS * S_LEN * S_LEN;  // [2]

  // workspace carve-up
  float* x    = (float*)d_ws;
  float* qkv  = x    + S_LEN * H_DIM;
  float* attn = qkv  + S_LEN * 3 * H_DIM;
  float* t2   = attn + S_LEN * H_DIM;
  float* ffh  = t2   + S_LEN * H_DIM;
  float* pooled = ffh + S_LEN * I_DIM;
  float* rstate = pooled + H_DIM;
  float* gi   = rstate + H_DIM;
  float* gh   = gi + 3 * H_DIM;
  float* p1v  = gh + 3 * H_DIM;
  bf16_t* tmpT   = (bf16_t*)(p1v + H_DIM);                // [S,H,H]=[i,o,y] bf16
  bf16_t* interB = tmpT + (size_t)S_LEN * H_DIM * H_DIM;  // [S,S,H]=[i,j,o] bf16

  dim3 blk(256);
  auto gemmStd = [&](const float* A, const float* B, const float* bias, float* C,
                     int M, int N, int K, bool gelu) {
    dim3 grd(M / 64, N / 64, 1);
    if (gelu)
      gemm_wmma_k<0, 0, 1, 0, 0, 0><<<grd, blk, 0, stream>>>(
          A, K, B, K, 0, bias, C, N, 0, K);
    else
      gemm_wmma_k<0, 0, 0, 0, 0, 0><<<grd, blk, 0, stream>>>(
          A, K, B, K, 0, bias, C, N, 0, K);
  };

  copy_k<<<dim3((S_LEN * H_DIM + 255) / 256), blk, 0, stream>>>(x, hidden, S_LEN * H_DIM);
  zero_k<<<dim3(2), blk, 0, stream>>>(rstate, H_DIM);

  for (int step = 0; step < NSTEPS; ++step) {
    // -------- transformer layers --------
    for (int l = 0; l < NLAYER; ++l) {
      const float* Wq = Wqkv + (size_t)l * 3 * H_DIM * H_DIM;
      gemmStd(x, Wq, bqkv + (size_t)l * 3 * H_DIM, qkv, S_LEN, 3 * H_DIM, H_DIM, false);
      attention_k<<<dim3(S_LEN, NHEAD), blk, 0, stream>>>(qkv, attn);
      gemmStd(attn, Wo + (size_t)l * H_DIM * H_DIM, bo + (size_t)l * H_DIM, t2,
              S_LEN, H_DIM, H_DIM, false);
      add_ln_k<<<dim3(S_LEN), blk, 0, stream>>>(x, t2, g1 + (size_t)l * H_DIM,
                                                be1 + (size_t)l * H_DIM);
      gemmStd(x, W1 + (size_t)l * I_DIM * H_DIM, b1 + (size_t)l * I_DIM, ffh,
              S_LEN, I_DIM, H_DIM, true);
      gemmStd(ffh, W2 + (size_t)l * H_DIM * I_DIM, b2 + (size_t)l * H_DIM, t2,
              S_LEN, H_DIM, I_DIM, false);
      add_ln_k<<<dim3(S_LEN), blk, 0, stream>>>(x, t2, g2 + (size_t)l * H_DIM,
                                                be2 + (size_t)l * H_DIM);
    }
    // -------- contradiction detector --------
    // tmp[i,o,y] = sum_x x[i,x] * Wbil[o,x,y]  (batched over o, B is K-major)
    {
      dim3 grd(S_LEN / 64, H_DIM / 64, H_DIM);
      gemm_wmma_k<1, 0, 0, 1, 0, 0><<<grd, blk, 0, stream>>>(
          x, H_DIM, Wbil, H_DIM, (long long)H_DIM * H_DIM, nullptr,
          tmpT, (long long)H_DIM * H_DIM, H_DIM, H_DIM);
    }
    // inter[i,j,o] = sum_y tmp[i,o,y] * x[j,y] + bbil[o]  (scatter store [i,j,o])
    {
      dim3 grd((S_LEN * H_DIM) / 64, S_LEN / 64, 1);
      gemm_wmma_k<0, 1, 0, 1, 1, 1><<<grd, blk, 0, stream>>>(
          tmpT, H_DIM, x, H_DIM, 0, bbil,
          interB, 0, 0, H_DIM);
    }
    // probs[i,j] = sigmoid(gelu(inter@Wc1^T + bc1) . Wc2 + bc2)  (fused)
    contra_head_k<<<dim3((S_LEN * S_LEN) / 128), blk, 0, stream>>>(
        interB, Wc1, bc1, Wc2, bc2, outCprob + (size_t)step * S_LEN * S_LEN);

    // -------- GRU controller on mean-pooled states --------
    meanpool_k<<<dim3(2), blk, 0, stream>>>(x, pooled);
    gemv_k<<<dim3((3 * H_DIM + 255) / 256), blk, 0, stream>>>(pooled, Wih, bih, gi,
                                                              3 * H_DIM, H_DIM, 0);
    gemv_k<<<dim3((3 * H_DIM + 255) / 256), blk, 0, stream>>>(rstate, Whh, bhh, gh,
                                                              3 * H_DIM, H_DIM, 0);
    gru_k<<<dim3(2), blk, 0, stream>>>(rstate, gi, gh);
    gemv_k<<<dim3(2), blk, 0, stream>>>(rstate, Wp1, bp1, p1v, H_DIM, H_DIM, 1);
    dotsig_k<<<dim3(1), blk, 0, stream>>>(p1v, Wp2, bp2, outConts + step);
  }
  // final projection
  gemmStd(x, Wout, bout, outFinal, S_LEN, H_DIM, H_DIM, false);
}
